// CauseFormer_83872121356647
// MI455X (gfx1250) — matrix operand order repaired and checked by reading, not compiled, not run on hardware
//
#include <hip/hip_runtime.h>
#include <hip/hip_bf16.h>
#include <cstdint>

// ---------------------------------------------------------------------------
// Problem constants (match reference)
// ---------------------------------------------------------------------------
#define BB   256
#define TT   200
#define DD   256
#define HH   4
#define DK   64          // DD / HH
#define LL   4
#define BT   (BB * TT)   // 51200 rows
#define TPAD 224         // TT padded to multiple of 32
#define NTIL 13          // ceil(TT/16)

typedef __attribute__((ext_vector_type(16))) __bf16 v16bf;
typedef __attribute__((ext_vector_type(8)))  __bf16 v8bf;
typedef __attribute__((ext_vector_type(8)))  float  v8f;
typedef int v4i_t __attribute__((vector_size(16)));  // matches async-builtin param

#ifndef __has_builtin
#define __has_builtin(x) 0
#endif

// ---------------------------------------------------------------------------
// CDNA5 async copy: GLOBAL_LOAD_ASYNC_TO_LDS_B128 (ASYNCcnt-tracked), with
// s_wait_asynccnt. Builtin if available, else inline asm per ISA 15.18.3.
// ---------------------------------------------------------------------------
__device__ __forceinline__ void async_ld128(const void* g, void* l) {
#if defined(__HIP_DEVICE_COMPILE__)
#if __has_builtin(__builtin_amdgcn_global_load_async_to_lds_b128)
  __builtin_amdgcn_global_load_async_to_lds_b128(
      (__attribute__((address_space(1))) v4i_t*)(g),
      (__attribute__((address_space(3))) v4i_t*)(l), 0, 0);
#else
  unsigned loff = (unsigned)(uintptr_t)(__attribute__((address_space(3))) char*)(l);
  asm volatile("global_load_async_to_lds_b128 %0, %1, off"
               :: "v"(loff), "v"(g) : "memory");
#endif
#endif
}

__device__ __forceinline__ void wait_async0() {
#if defined(__HIP_DEVICE_COMPILE__)
#if __has_builtin(__builtin_amdgcn_s_wait_asynccnt)
  __builtin_amdgcn_s_wait_asynccnt((short)0);
#else
  asm volatile("s_wait_asynccnt 0x0" ::: "memory");
#endif
#endif
}

__device__ __forceinline__ void wait_async6() {
#if defined(__HIP_DEVICE_COMPILE__)
#if __has_builtin(__builtin_amdgcn_s_wait_asynccnt)
  __builtin_amdgcn_s_wait_asynccnt((short)6);
#else
  asm volatile("s_wait_asynccnt 0x6" ::: "memory");
#endif
#endif
}

// ---------------------------------------------------------------------------
// Fragment helpers (ISA 7.12.2 layouts)
//  A 16x32 bf16: lane L(0-15) row M=L holds K={0..7,16..23}; lane L+16 holds
//  K={8..15,24..31}.  B 32x16: lanes 0-15 col N=lane K={0..7,16..23};
//  lanes 16-31 col N=lane-16 K={8..15,24..31}.
// ---------------------------------------------------------------------------
__device__ __forceinline__ v16bf combine16(v8bf lo, v8bf hi) {
  v16bf a;
#pragma unroll
  for (int j = 0; j < 8; ++j) { a[j] = lo[j]; a[8 + j] = hi[j]; }
  return a;
}

__device__ __forceinline__ v16bf afrag_bf16(const __bf16* rowp, int half) {
  // rowp points at the K-chunk base of this row (LDS or global)
  v8bf lo = *(const v8bf*)(rowp + half * 8);
  v8bf hi = *(const v8bf*)(rowp + 16 + half * 8);
  return combine16(lo, hi);
}

__device__ __forceinline__ v16bf bfrag_bf16(const __bf16* colp, int half) {
  // colp points at the K-chunk base of this column's contiguous-K storage
  const __bf16* p = colp + half * 8;
  v8bf lo = *(const v8bf*)(p);
  v8bf hi = *(const v8bf*)(p + 16);
  return combine16(lo, hi);
}

#define WMMA_BF16(A, Bm, C) \
  __builtin_amdgcn_wmma_f32_16x16x32_bf16(false, (A), false, (Bm), (short)0, (C), false, false)

// ---------------------------------------------------------------------------
// Weight prep: WT[li][n][k] = (bf16) W[li][k][n]   (once per launch)
// ---------------------------------------------------------------------------
__global__ __launch_bounds__(256) void wprep_kernel(const float* __restrict__ W,
                                                    __bf16* __restrict__ WT) {
  size_t idx = (size_t)blockIdx.x * 256 + threadIdx.x;
  if (idx >= (size_t)LL * DD * DD) return;
  int li = (int)(idx / (DD * DD));
  int n  = (int)((idx / DD) % DD);
  int k  = (int)(idx % DD);
  WT[idx] = (__bf16)W[(size_t)li * DD * DD + (size_t)k * DD + n];
}

// ---------------------------------------------------------------------------
// Embedding: fp32 + bf16 outputs
// ---------------------------------------------------------------------------
__global__ __launch_bounds__(256) void embed_kernel(const int* __restrict__ logs,
                                                    const float* __restrict__ itab,
                                                    const float* __restrict__ ptab,
                                                    float* __restrict__ outF,
                                                    __bf16* __restrict__ outB) {
  size_t idx = (size_t)blockIdx.x * 256 + threadIdx.x;
  if (idx >= (size_t)BT * DD) return;
  int row = (int)(idx / DD);
  int d   = (int)(idx % DD);
  int t   = row % TT;
  int item = logs[row];
  int pos  = (item != 0) ? (t + 1) : 0;
  float v = itab[(size_t)item * DD + d] * 16.0f + ptab[(size_t)pos * DD + d];
  outF[idx] = v;
  outB[idx] = (__bf16)v;
}

// ---------------------------------------------------------------------------
// LayerNorm (wave per row), optional bf16 twin output
// ---------------------------------------------------------------------------
__global__ __launch_bounds__(256) void ln_kernel(const float* __restrict__ x,
                                                 const float* __restrict__ sc,
                                                 const float* __restrict__ bs,
                                                 float* __restrict__ outF,
                                                 __bf16* __restrict__ outB, int nrows) {
  int wave = threadIdx.x >> 5, lane = threadIdx.x & 31;
  int row = blockIdx.x * 8 + wave;
  if (row >= nrows) return;
  const float* xr = x + (size_t)row * DD;
  float vals[8];
  const float4 v0 = *(const float4*)(xr + lane * 8);
  const float4 v1 = *(const float4*)(xr + lane * 8 + 4);
  vals[0] = v0.x; vals[1] = v0.y; vals[2] = v0.z; vals[3] = v0.w;
  vals[4] = v1.x; vals[5] = v1.y; vals[6] = v1.z; vals[7] = v1.w;
  float m = 0.f;
#pragma unroll
  for (int j = 0; j < 8; ++j) m += vals[j];
#pragma unroll
  for (int o = 16; o > 0; o >>= 1) m += __shfl_xor(m, o, 32);
  m *= (1.0f / DD);
  float var = 0.f;
#pragma unroll
  for (int j = 0; j < 8; ++j) { float d = vals[j] - m; var += d * d; }
#pragma unroll
  for (int o = 16; o > 0; o >>= 1) var += __shfl_xor(var, o, 32);
  var *= (1.0f / DD);
  float inv = 1.0f / sqrtf(var + 1e-8f);
#pragma unroll
  for (int j = 0; j < 8; ++j) {
    int d = lane * 8 + j;
    float v = (vals[j] - m) * inv * sc[d] + bs[d];
    outF[(size_t)row * DD + d] = v;
    if (outB) outB[(size_t)row * DD + d] = (__bf16)v;
  }
}

// Row mean over D (for covariance mu)
__global__ __launch_bounds__(256) void row_mean_kernel(const float* __restrict__ x,
                                                       float* __restrict__ mu, int nrows) {
  int wave = threadIdx.x >> 5, lane = threadIdx.x & 31;
  int row = blockIdx.x * 8 + wave;
  if (row >= nrows) return;
  const float* xr = x + (size_t)row * DD;
  float m = 0.f;
#pragma unroll
  for (int j = 0; j < 8; ++j) m += xr[lane * 8 + j];
#pragma unroll
  for (int o = 16; o > 0; o >>= 1) m += __shfl_xor(m, o, 32);
  if (lane == 0) mu[row] = m * (1.0f / DD);
}

// ---------------------------------------------------------------------------
// GEMM: out = act( A[M,K](bf16) @ W + bias + res ), weights given transposed
// bf16 WT[N,K]. Async double-buffered LDS staging, K-step 64, tile 128x64.
// ---------------------------------------------------------------------------
#define GTM 128
#define GTN 64
#define GTK2 64
#define LDW 72   // LDS K-stride (bf16): 144B rows -> conflict-free b128 reads

__global__ __launch_bounds__(256)
void gemm_async_kernel(const __bf16* __restrict__ A, const __bf16* __restrict__ WT,
                       const float* __restrict__ bias, const float* __restrict__ res,
                       float* __restrict__ outF, __bf16* __restrict__ outB,
                       int M, int N, int K, int relu) {
  __shared__ __bf16 As[2][GTM][LDW];
  __shared__ __bf16 Bs[2][GTN][LDW];

  const int tid = threadIdx.x;
  const int lane = tid & 31, wave = tid >> 5;
  const int wm = (wave & 3) * 32, wn = (wave >> 2) * 32;
  const int mBase = blockIdx.x * GTM, nBase = blockIdx.y * GTN;
  const int half = lane >> 4, l15 = lane & 15;
  const int nstage = K / GTK2;

  auto issueStage = [&](int ks, int buf) {
    const int kk = ks * GTK2;
    // A tile: 128 rows x 8 x16B chunks = 1024 chunks -> 4 per thread
#pragma unroll
    for (int i = 0; i < 4; ++i) {
      int idx = tid + i * 256;
      int r = idx >> 3, c = idx & 7;
      async_ld128(A + (size_t)(mBase + r) * K + kk + c * 8, &As[buf][r][c * 8]);
    }
    // W tile: 64 rows x 8 chunks = 512 -> 2 per thread
#pragma unroll
    for (int i = 0; i < 2; ++i) {
      int idx = tid + i * 256;
      int r = idx >> 3, c = idx & 7;
      async_ld128(WT + (size_t)(nBase + r) * K + kk + c * 8, &Bs[buf][r][c * 8]);
    }
  };

  v8f acc[2][2];
#pragma unroll
  for (int i = 0; i < 2; ++i)
#pragma unroll
    for (int j = 0; j < 2; ++j)
#pragma unroll
      for (int r = 0; r < 8; ++r) acc[i][j][r] = 0.f;

  issueStage(0, 0);
  for (int ks = 0; ks < nstage; ++ks) {
    const int buf = ks & 1;
    if (ks + 1 < nstage) { issueStage(ks + 1, buf ^ 1); wait_async6(); }
    else                 { wait_async0(); }
    __syncthreads();
#pragma unroll
    for (int kc = 0; kc < GTK2; kc += 32) {
      v16bf af[2], bf[2];
#pragma unroll
      for (int sm = 0; sm < 2; ++sm)
        af[sm] = afrag_bf16(&As[buf][wm + sm * 16 + l15][kc], half);
#pragma unroll
      for (int sn = 0; sn < 2; ++sn)
        bf[sn] = bfrag_bf16(&Bs[buf][wn + sn * 16 + l15][kc], half);
#pragma unroll
      for (int sm = 0; sm < 2; ++sm)
#pragma unroll
        for (int sn = 0; sn < 2; ++sn)
          acc[sm][sn] = WMMA_BF16(af[sm], bf[sn], acc[sm][sn]);
    }
    __syncthreads();
  }

  // epilogue: element (M = r + 8*half, N = l15)
#pragma unroll
  for (int sm = 0; sm < 2; ++sm)
#pragma unroll
    for (int sn = 0; sn < 2; ++sn) {
      int n = nBase + wn + sn * 16 + l15;
      float bv = bias[n];
#pragma unroll
      for (int r = 0; r < 8; ++r) {
        int m = mBase + wm + sm * 16 + r + half * 8;
        if (m < M) {
          float v = acc[sm][sn][r] + bv;
          if (res) v += res[(size_t)m * N + n];
          if (relu) v = fmaxf(v, 0.f);
          if (outF) outF[(size_t)m * N + n] = v;
          if (outB) outB[(size_t)m * N + n] = (__bf16)v;
        }
      }
    }
}

// ---------------------------------------------------------------------------
// Attention: one wave per (b,h,q-tile); bf16 q/k/v in, bf16 o out.
// ---------------------------------------------------------------------------
__global__ __launch_bounds__(32)
void attn_kernel(const __bf16* __restrict__ q, const __bf16* __restrict__ k,
                 const __bf16* __restrict__ v, const float* __restrict__ cov,
                 __bf16* __restrict__ o, int useCov) {
  __shared__ float  S[16][TPAD];
  __shared__ __bf16 P[16][TPAD];
  __shared__ __bf16 VT[DK][32];

  const int lane = threadIdx.x;
  const int qt = blockIdx.x, h = blockIdx.y, b = blockIdx.z;
  const int half = lane >> 4, l15 = lane & 15;

  const int mload = qt * 16 + l15;
  const __bf16* qrow = q + ((size_t)(b * TT + (mload < TT ? mload : 0)) * DD + h * DK);
  v16bf qf[2];
#pragma unroll
  for (int c = 0; c < 2; ++c) qf[c] = afrag_bf16(qrow + c * 32, half);

  for (int kt = 0; kt < NTIL; ++kt) {
    const int nload = kt * 16 + l15;
    const __bf16* krow = k + ((size_t)(b * TT + (nload < TT ? nload : 0)) * DD + h * DK);
    v8f accs;
#pragma unroll
    for (int r = 0; r < 8; ++r) accs[r] = 0.f;
#pragma unroll
    for (int c = 0; c < 2; ++c) {
      v16bf kf = bfrag_bf16(krow + c * 32, half);
      accs = WMMA_BF16(qf[c], kf, accs);
    }
#pragma unroll
    for (int r = 0; r < 8; ++r) {
      int m = qt * 16 + r + half * 8;
      int n = kt * 16 + l15;
      float sc = accs[r] * 0.125f;              // 1/sqrt(dk)
      if (n > m) sc = -1e9f;                    // causal mask (triu k=1)
      if (useCov && m < TT && n < TT)
        sc *= (1.0f + 0.5f * cov[((size_t)b * TT + m) * TT + n]);
      if (n >= TT || m >= TT) sc = -1e9f;       // padding
      S[r + half * 8][n] = sc;
    }
  }
  for (int i = lane; i < 16 * (TPAD - NTIL * 16); i += 32) {
    int r = i / (TPAD - NTIL * 16), c = i % (TPAD - NTIL * 16);
    S[r][NTIL * 16 + c] = -1e9f;
  }
  __syncthreads();

  {  // softmax: 2 lanes per row, 112 cols each
    int row = lane >> 1, seg = lane & 1;
    int c0 = seg * 112;
    float mx = -3.4e38f;
    for (int c = c0; c < c0 + 112; ++c) mx = fmaxf(mx, S[row][c]);
    mx = fmaxf(mx, __shfl_xor(mx, 1, 32));
    float sum = 0.f;
    for (int c = c0; c < c0 + 112; ++c) {
      float e = __expf(S[row][c] - mx);
      S[row][c] = e;
      sum += e;
    }
    sum += __shfl_xor(sum, 1, 32);
    float inv = 1.0f / sum;
    for (int c = c0; c < c0 + 112; ++c) P[row][c] = (__bf16)(S[row][c] * inv);
  }
  __syncthreads();

  v8f oacc[4];
#pragma unroll
  for (int nt = 0; nt < 4; ++nt)
#pragma unroll
    for (int r = 0; r < 8; ++r) oacc[nt][r] = 0.f;

  for (int ks = 0; ks < TPAD / 32; ++ks) {
    {  // stage V transposed: VT[d][kvl]
      int kv = ks * 32 + lane;
      const __bf16* vrow = v + ((size_t)(b * TT + (kv < TT ? kv : 0)) * DD + h * DK);
#pragma unroll
      for (int d = 0; d < DK; ++d) VT[d][lane] = vrow[d];
    }
    __syncthreads();
    v16bf pf = afrag_bf16(&P[l15][ks * 32], half);
#pragma unroll
    for (int nt = 0; nt < 4; ++nt) {
      v16bf vf = bfrag_bf16(&VT[nt * 16 + l15][0], half);
      oacc[nt] = WMMA_BF16(pf, vf, oacc[nt]);
    }
    __syncthreads();
  }

#pragma unroll
  for (int nt = 0; nt < 4; ++nt)
#pragma unroll
    for (int r = 0; r < 8; ++r) {
      int m = qt * 16 + r + half * 8;
      if (m < TT)
        o[((size_t)(b * TT + m) * DD + h * DK + nt * 16 + l15)] = (__bf16)oacc[nt][r];
    }
}

// ---------------------------------------------------------------------------
// Covariance: cov[b,t,s] = min((Z_t.Z_s - D*mu_t*mu_s)/(D-1), 3); z is bf16
// ---------------------------------------------------------------------------
__global__ __launch_bounds__(32)
void cov_kernel(const __bf16* __restrict__ z, const float* __restrict__ mu,
                float* __restrict__ cov) {
  const int lane = threadIdx.x;
  const int tt = blockIdx.x, st = blockIdx.y, b = blockIdx.z;
  const int half = lane >> 4, l15 = lane & 15;

  int tload = tt * 16 + l15; if (tload >= TT) tload = 0;
  int sload = st * 16 + l15; if (sload >= TT) sload = 0;
  const __bf16* arow = z + (size_t)(b * TT + tload) * DD;
  const __bf16* brow = z + (size_t)(b * TT + sload) * DD;

  v8f acc;
#pragma unroll
  for (int r = 0; r < 8; ++r) acc[r] = 0.f;
#pragma unroll
  for (int c = 0; c < DD / 32; ++c) {
    v16bf af = afrag_bf16(arow + c * 32, half);
    v16bf bf = bfrag_bf16(brow + c * 32, half);
    acc = WMMA_BF16(af, bf, acc);
  }
#pragma unroll
  for (int r = 0; r < 8; ++r) {
    int t = tt * 16 + r + half * 8;
    int s = st * 16 + l15;
    if (t < TT && s < TT) {
      float g = (acc[r] - (float)DD * mu[b * TT + t] * mu[b * TT + s]) * (1.0f / (DD - 1));
      cov[((size_t)b * TT + t) * TT + s] = (g < 3.0f) ? g : 3.0f;
    }
  }
}

// ---------------------------------------------------------------------------
// Host orchestration
// ---------------------------------------------------------------------------
extern "C" void kernel_launch(void* const* d_in, const int* in_sizes, int n_in,
                              void* d_out, int out_size, void* d_ws, size_t ws_size,
                              hipStream_t stream) {
  (void)in_sizes; (void)n_in; (void)out_size; (void)ws_size;
  const int*   log_seqs = (const int*)d_in[0];
  const float* item_tab = (const float*)d_in[1];
  const float* pos_tab  = (const float*)d_in[2];
  const float* ln1_s = (const float*)d_in[3];
  const float* ln1_b = (const float*)d_in[4];
  const float* Wq = (const float*)d_in[5];
  const float* bq = (const float*)d_in[6];
  const float* Wk = (const float*)d_in[7];
  const float* bk = (const float*)d_in[8];
  const float* Wv = (const float*)d_in[9];
  const float* bv = (const float*)d_in[10];
  const float* Wo = (const float*)d_in[11];
  const float* bo = (const float*)d_in[12];
  const float* ln2_s = (const float*)d_in[13];
  const float* ln2_b = (const float*)d_in[14];
  const float* W1 = (const float*)d_in[15];
  const float* b1 = (const float*)d_in[16];
  const float* W2 = (const float*)d_in[17];
  const float* b2 = (const float*)d_in[18];
  const float* last_s = (const float*)d_in[19];
  const float* last_b = (const float*)d_in[20];

  // --- workspace carving (all 16B-aligned) ---
  const size_t EF = (size_t)BT * DD;
  char* p = (char*)d_ws;
  float* F[6];
  for (int i = 0; i < 6; ++i) { F[i] = (float*)p; p += EF * 4; }
  __bf16* Hb[8];
  for (int i = 0; i < 8; ++i) { Hb[i] = (__bf16*)p; p += EF * 2; }
  float* covb = (float*)p; p += (size_t)BB * TT * TT * 4;
  float* mub  = (float*)p; p += (size_t)BT * 4;
  __bf16* WTs[6];
  for (int i = 0; i < 6; ++i) { WTs[i] = (__bf16*)p; p += (size_t)LL * DD * DD * 2; }
  // F: 0=seqs 1=normQ 2=s 3=y 4=z-ping 5=z-pong
  // Hb: 0=seqs 1=normQ 2=q/y 3=k/h 4=v 5=o 6=z-ping 7=z-pong
  // WTs: 0=Wq 1=Wk 2=Wv 3=Wo 4=W1 5=W2 (bf16, transposed [L][n][k])

  const dim3 gemmGrid(BT / GTM, DD / GTN);  // (400, 4)
  const dim3 attnGrid(NTIL, HH, BB);
  const dim3 covGrid(NTIL, NTIL, BB);
  const int lnBlocks = BT / 8;
  const int wBlocks = (LL * DD * DD) / 256;  // 1024

  // weight transposition+conversion (once)
  wprep_kernel<<<wBlocks, 256, 0, stream>>>(Wq, WTs[0]);
  wprep_kernel<<<wBlocks, 256, 0, stream>>>(Wk, WTs[1]);
  wprep_kernel<<<wBlocks, 256, 0, stream>>>(Wv, WTs[2]);
  wprep_kernel<<<wBlocks, 256, 0, stream>>>(Wo, WTs[3]);
  wprep_kernel<<<wBlocks, 256, 0, stream>>>(W1, WTs[4]);
  wprep_kernel<<<wBlocks, 256, 0, stream>>>(W2, WTs[5]);

  embed_kernel<<<(unsigned)((EF + 255) / 256), 256, 0, stream>>>(log_seqs, item_tab,
                                                                 pos_tab, F[0], Hb[0]);

  auto encoder = [&](const float* xf, const __bf16* xb, float* of, __bf16* ob, int li,
                     int useCov) {
    const float* l1s = ln1_s + li * DD;  const float* l1b = ln1_b + li * DD;
    const float* l2s = ln2_s + li * DD;  const float* l2b = ln2_b + li * DD;
    const size_t wo_ = (size_t)li * DD * DD;
    // Q = LN1(x)
    ln_kernel<<<lnBlocks, 256, 0, stream>>>(xf, l1s, l1b, F[1], Hb[1], BT);
    // q from normQ; k,v from raw x  (bf16-only outputs)
    gemm_async_kernel<<<gemmGrid, 256, 0, stream>>>(Hb[1], WTs[0] + wo_, bq + li * DD,
                                                    nullptr, nullptr, Hb[2], BT, DD, DD, 0);
    gemm_async_kernel<<<gemmGrid, 256, 0, stream>>>(xb, WTs[1] + wo_, bk + li * DD,
                                                    nullptr, nullptr, Hb[3], BT, DD, DD, 0);
    gemm_async_kernel<<<gemmGrid, 256, 0, stream>>>(xb, WTs[2] + wo_, bv + li * DD,
                                                    nullptr, nullptr, Hb[4], BT, DD, DD, 0);
    attn_kernel<<<attnGrid, 32, 0, stream>>>(Hb[2], Hb[3], Hb[4], covb, Hb[5], useCov);
    // s = normQ + (o @ Wo + bo)
    gemm_async_kernel<<<gemmGrid, 256, 0, stream>>>(Hb[5], WTs[3] + wo_, bo + li * DD,
                                                    F[1], F[2], nullptr, BT, DD, DD, 0);
    // y = LN2(s)
    ln_kernel<<<lnBlocks, 256, 0, stream>>>(F[2], l2s, l2b, F[3], Hb[2], BT);
    // h = relu(y @ W1 + b1)
    gemm_async_kernel<<<gemmGrid, 256, 0, stream>>>(Hb[2], WTs[4] + wo_, b1 + li * DD,
                                                    nullptr, nullptr, Hb[3], BT, DD, DD, 1);
    // out = h @ W2 + b2 + y
    gemm_async_kernel<<<gemmGrid, 256, 0, stream>>>(Hb[3], WTs[5] + wo_, b2 + li * DD,
                                                    F[3], of, ob, BT, DD, DD, 0);
  };

  // Causal layers all re-read the original seqs; only index L//2-1 contributes.
  encoder(F[0], Hb[0], F[4], Hb[6], LL / 2 - 1, 0);
  row_mean_kernel<<<lnBlocks, 256, 0, stream>>>(F[4], mub, BT);
  cov_kernel<<<covGrid, 32, 0, stream>>>(Hb[6], mub, covb);

  // Decoder layers with multiplicative cov reweighting
  encoder(F[0], Hb[0], F[5], Hb[7], 2, 1);
  encoder(F[5], Hb[7], F[4], Hb[6], 3, 1);

  // Final LayerNorm -> d_out (fp32)
  ln_kernel<<<lnBlocks, 256, 0, stream>>>(F[4], last_s, last_b, (float*)d_out, nullptr, BT);
}
